// ConvLSTMBlock_32461362823758
// MI455X (gfx1250) — matrix operand order repaired and checked by reading, not compiled, not run on hardware
//
#include <hip/hip_runtime.h>
#include <hip/hip_bf16.h>
#include <math.h>

// ---------------------------------------------------------------------------
// ConvLSTM2D (Keras-style) + fused inference BatchNorm for MI455X / gfx1250.
// Implicit-GEMM convs via v_wmma_f32_16x16x32_bf16 (bf16 operands, f32 acc).
// Per timestep: z = conv(x_t, W, s2) + conv(h, U, s1) + b, fused gates/BN.
//
// Tiling: one workgroup = one output row (M=64 spatial) x all 256 cout.
// Wave w: M-tiles {2p, 2p+1} (p = w>>2), feature slice fb = (w&3)*16, and all
// four gate column groups (cout = g*64 + fb + lane) -> 8 accumulators.
// Each B fragment feeds 2 WMMAs, each A fragment feeds 4 WMMAs, and the four
// gate accumulators are elementwise-aligned so the LSTM cell update runs
// straight out of the WMMA C/D register layout (no LDS z-exchange).
// ---------------------------------------------------------------------------

typedef __bf16 bf16;
typedef __attribute__((ext_vector_type(16))) __bf16 v16bf;
typedef __attribute__((ext_vector_type(8)))  float  v8f;
typedef __attribute__((ext_vector_type(4)))  unsigned int u32x4;

union Frag32 { v16bf v; u32x4 q[2]; };   // 32 bytes: one 16-elem bf16 operand

#define NB   4      // batch
#define NT   16     // timesteps
#define HIN  128
#define WINW 128
#define CIN  32
#define NF   64     // features F
#define H2   64
#define W2   64
#define C4F  256    // 4*F

// workspace layout (bytes)
#define OFF_WT    0u              // 9*256*32  bf16 = 147456
#define OFF_UT    147456u         // 9*256*64  bf16 = 294912
#define OFF_SCALE 442368u         // 64 f32
#define OFF_SHIFT 442624u         // 64 f32
#define OFF_H0    442880u         // 4*64*64*64 bf16 = 2097152
#define OFF_H1    2540032u        // 4*64*64*64 bf16 = 2097152
#define OFF_C     4637184u        // 4*64*64*64 f32  = 4194304  (end 8831488)

// ---------------------------------------------------------------------------
// Prep: transpose+convert weights to bf16 [tap][cout][cin]; fold BN stats.
// ---------------------------------------------------------------------------
__global__ void convlstm_prep(const float* __restrict__ W,
                              const float* __restrict__ U,
                              const float* __restrict__ gamma,
                              const float* __restrict__ beta,
                              const float* __restrict__ mmean,
                              const float* __restrict__ mvar,
                              bf16* __restrict__ Wt,
                              bf16* __restrict__ Ut,
                              float* __restrict__ scale,
                              float* __restrict__ shift)
{
    int idx = blockIdx.x * blockDim.x + threadIdx.x;
    int stride = gridDim.x * blockDim.x;
    // Wt[tap][co][ci] = W[tap][ci][co]
    for (int i = idx; i < 9 * C4F * CIN; i += stride) {
        int ci = i % CIN; int co = (i / CIN) % C4F; int tap = i / (CIN * C4F);
        Wt[i] = (bf16)W[(tap * CIN + ci) * C4F + co];
    }
    // Ut[tap][co][ci] = U[tap][ci][co]
    for (int i = idx; i < 9 * C4F * NF; i += stride) {
        int ci = i % NF; int co = (i / NF) % C4F; int tap = i / (NF * C4F);
        Ut[i] = (bf16)U[(tap * NF + ci) * C4F + co];
    }
    if (idx < NF) {
        float s = gamma[idx] * rsqrtf(mvar[idx] + 1e-3f);
        scale[idx] = s;
        shift[idx] = beta[idx] - mmean[idx] * s;
    }
}

// ---------------------------------------------------------------------------
// Zero h0 and c state (must run every launch: harness poisons ws once).
// ---------------------------------------------------------------------------
__global__ void convlstm_zero(unsigned int* __restrict__ h0w,
                              unsigned int* __restrict__ cw)
{
    int idx = blockIdx.x * blockDim.x + threadIdx.x;
    int stride = gridDim.x * blockDim.x;
    for (int i = idx; i < NB * H2 * W2 * NF / 2; i += stride) h0w[i] = 0u; // bf16 pairs
    for (int i = idx; i < NB * H2 * W2 * NF; i += stride)     cw[i]  = 0u; // f32 zero
}

// ---------------------------------------------------------------------------
// One recurrence step. Grid (64, 4) = (y, batch). 256 threads = 8 waves.
// ---------------------------------------------------------------------------
__global__ __launch_bounds__(256)
void convlstm_step(const float* __restrict__ x,
                   const bf16*  __restrict__ Wt,
                   const bf16*  __restrict__ Ut,
                   const float* __restrict__ bias,
                   const float* __restrict__ scale,
                   const float* __restrict__ shift,
                   const bf16*  __restrict__ hin,
                   bf16*        __restrict__ hout,
                   float*       __restrict__ cst,
                   float*       __restrict__ out,
                   int t)
{
    __shared__ bf16 sX[3][130][32];   // stride-2 input rows 2y-1..2y+1, cols -1..128
    __shared__ bf16 sH[3][66][64];    // h rows y-1..y+1, cols -1..64

    const int tid = threadIdx.x;
    const int y   = blockIdx.x;
    const int b   = blockIdx.y;

    // ---- stage x tile (bf16, zero-padded SAME); tile col j <-> input col j-1
    for (int i = tid; i < 3 * 130 * 32; i += 256) {
        int ci = i & 31;
        int j  = (i >> 5) % 130;
        int ky = (i >> 5) / 130;
        int r  = 2 * y - 1 + ky;
        int c  = j - 1;
        float v = 0.f;
        if ((unsigned)r < HIN && (unsigned)c < WINW)
            v = x[(((size_t)(b * NT + t) * HIN + r) * WINW + c) * CIN + ci];
        sX[ky][j][ci] = (bf16)v;
    }
    // ---- stage h tile; tile col j <-> h col j-1
    for (int i = tid; i < 3 * 66 * 64; i += 256) {
        int ci = i & 63;
        int j  = (i >> 6) % 66;
        int ky = (i >> 6) / 66;
        int r  = y - 1 + ky;
        int c  = j - 1;
        bf16 v = (bf16)0.f;
        if ((unsigned)r < H2 && (unsigned)c < W2)
            v = hin[(((size_t)b * H2 + r) * W2 + c) * NF + ci];
        sH[ky][j][ci] = v;
    }
    __syncthreads();

    const int wave = tid >> 5;
    const int lane = tid & 31;
    const int hi   = lane >> 4;        // half-wave select
    const int lcol = lane & 15;
    const int p2   = (wave >> 2) * 2;  // first of this wave's two M-tiles
    const int fb   = (wave & 3) * 16;  // feature slice base (within a gate)
    const int dA   = hi * 8;           // A-layout cin sub-offset
    const int dB   = hi * 8;           // B-layout cin sub-offset
    const int fch  = fb + lcol;        // this lane's feature channel

    v8f acc[2][4] = {};                // [m-tile][gate i,f,g,o]

    // ================= input conv, stride 2 : 9 taps x 32 cin ==============
    for (int tap = 0; tap < 9; ++tap) {
        const int kx = tap % 3, ky = tap / 3;
        const bf16* bp = Wt + (size_t)tap * C4F * CIN;
        __builtin_prefetch(bp + (size_t)C4F * CIN, 0, 1);  // next tap weights

        Frag32 a[2];
#pragma unroll
        for (int mt = 0; mt < 2; ++mt) {
            int gm = (p2 + mt) * 16 + lcol;                 // A row (spatial)
            const bf16* ap = &sX[ky][2 * gm + kx][0];
            a[mt].q[0] = *reinterpret_cast<const u32x4*>(ap + dA);
            a[mt].q[1] = *reinterpret_cast<const u32x4*>(ap + 16 + dA);
        }
        Frag32 bm[4];
#pragma unroll
        for (int g = 0; g < 4; ++g) {
            const bf16* p = bp + (size_t)(g * NF + fch) * CIN + dB;
            bm[g].q[0] = *reinterpret_cast<const u32x4*>(p);
            bm[g].q[1] = *reinterpret_cast<const u32x4*>(p + 16);
        }
#pragma unroll
        for (int mt = 0; mt < 2; ++mt)
#pragma unroll
            for (int g = 0; g < 4; ++g)
                acc[mt][g] = __builtin_amdgcn_wmma_f32_16x16x32_bf16(
                    false, a[mt].v, false, bm[g].v, (short)0, acc[mt][g],
                    false, false);
    }

    // ============ recurrent conv, stride 1 : 9 taps x 64 cin ===============
    for (int tap = 0; tap < 9; ++tap) {
        const int kx = tap % 3, ky = tap / 3;
        const bf16* up = Ut + (size_t)tap * C4F * NF;
        __builtin_prefetch(up + (size_t)C4F * NF, 0, 1);   // next tap weights
#pragma unroll
        for (int cb = 0; cb < NF; cb += 32) {
            Frag32 a[2];
#pragma unroll
            for (int mt = 0; mt < 2; ++mt) {
                int gm = (p2 + mt) * 16 + lcol;
                const bf16* hp = &sH[ky][gm + kx][0];
                a[mt].q[0] = *reinterpret_cast<const u32x4*>(hp + cb + dA);
                a[mt].q[1] = *reinterpret_cast<const u32x4*>(hp + cb + 16 + dA);
            }
            Frag32 bm[4];
#pragma unroll
            for (int g = 0; g < 4; ++g) {
                const bf16* p = up + (size_t)(g * NF + fch) * NF + cb + dB;
                bm[g].q[0] = *reinterpret_cast<const u32x4*>(p);
                bm[g].q[1] = *reinterpret_cast<const u32x4*>(p + 16);
            }
#pragma unroll
            for (int mt = 0; mt < 2; ++mt)
#pragma unroll
                for (int g = 0; g < 4; ++g)
                    acc[mt][g] = __builtin_amdgcn_wmma_f32_16x16x32_bf16(
                        false, a[mt].v, false, bm[g].v, (short)0, acc[mt][g],
                        false, false);
        }
    }

    // ---- gates (Keras i,f,g,o) straight from C/D register layout ----------
    // elem r of acc -> spatial row gx = (p2+mt)*16 + r + 8*hi, feature = fch.
    const float bi = bias[fch];
    const float bf_ = bias[64 + fch];
    const float bg = bias[128 + fch];
    const float bo = bias[192 + fch];
    const float sc = scale[fch];
    const float sh = shift[fch];

#pragma unroll
    for (int mt = 0; mt < 2; ++mt) {
#pragma unroll
        for (int r = 0; r < 8; ++r) {
            int gx = (p2 + mt) * 16 + r + hi * 8;
            float zi = acc[mt][0][r] + bi;
            float zf = acc[mt][1][r] + bf_;
            float zg = acc[mt][2][r] + bg;
            float zo = acc[mt][3][r] + bo;
            float gi = fminf(fmaxf(0.2f * zi + 0.5f, 0.f), 1.f);
            float gf = fminf(fmaxf(0.2f * zf + 0.5f, 0.f), 1.f);
            float go = fminf(fmaxf(0.2f * zo + 0.5f, 0.f), 1.f);
            size_t sidx = (((size_t)b * H2 + y) * W2 + gx) * NF + fch;
            float cold = cst[sidx];
            float cnew = gf * cold + gi * tanhf(zg);
            float h    = go * tanhf(cnew);
            cst[sidx]  = cnew;
            hout[sidx] = (bf16)h;
            out[((((size_t)b * NT + t) * H2 + y) * W2 + gx) * NF + fch]
                = h * sc + sh;
        }
    }
}

// ---------------------------------------------------------------------------
extern "C" void kernel_launch(void* const* d_in, const int* in_sizes, int n_in,
                              void* d_out, int out_size, void* d_ws, size_t ws_size,
                              hipStream_t stream)
{
    const float* x     = (const float*)d_in[0];
    const float* W     = (const float*)d_in[1];
    const float* U     = (const float*)d_in[2];
    const float* bias  = (const float*)d_in[3];
    const float* gamma = (const float*)d_in[4];
    const float* beta  = (const float*)d_in[5];
    const float* mmean = (const float*)d_in[6];
    const float* mvar  = (const float*)d_in[7];
    float* out = (float*)d_out;

    char* ws = (char*)d_ws;
    bf16*  Wt    = (bf16*) (ws + OFF_WT);
    bf16*  Ut    = (bf16*) (ws + OFF_UT);
    float* scale = (float*)(ws + OFF_SCALE);
    float* shift = (float*)(ws + OFF_SHIFT);
    bf16*  h0    = (bf16*) (ws + OFF_H0);
    bf16*  h1    = (bf16*) (ws + OFF_H1);
    float* cst   = (float*)(ws + OFF_C);

    convlstm_prep<<<256, 256, 0, stream>>>(W, U, gamma, beta, mmean, mvar,
                                           Wt, Ut, scale, shift);
    convlstm_zero<<<1024, 256, 0, stream>>>((unsigned int*)h0,
                                            (unsigned int*)cst);

    dim3 grid(64, 4);
    for (int t = 0; t < NT; ++t) {
        const bf16* hin  = (t & 1) ? h1 : h0;
        bf16*       hout = (t & 1) ? h0 : h1;
        convlstm_step<<<grid, 256, 0, stream>>>(x, Wt, Ut, bias, scale, shift,
                                                hin, hout, cst, out, t);
    }
}
